// E3EGNN_edge_63024350101882
// MI455X (gfx1250) — matrix operand order repaired and checked by reading, not compiled, not run on hardware
//
#include <hip/hip_runtime.h>
#include <hip/hip_bf16.h>

// ---------------------------------------------------------------------------
// CDNA5 (gfx1250) EGNN: edge/node MLPs on WMMA (v_wmma_f32_16x16x32_f16),
// gathers/scatter-atomics ride the 192MB L2. wave32: one 16-edge (or 16-node)
// tile per wave, 8 waves / 256-thread block. Bias folded into WMMA C operand;
// activations use v_exp_f32 + v_rcp_f32; no exec-mask diamonds in the hot path.
// ---------------------------------------------------------------------------

typedef __attribute__((ext_vector_type(16))) _Float16 v16h;
typedef __attribute__((ext_vector_type(8)))  float    v8f;

__device__ __forceinline__ float fast_rcp(float v) { return __builtin_amdgcn_rcpf(v); }
__device__ __forceinline__ float sigm_f(float v) { return fast_rcp(1.0f + __expf(-v)); }
__device__ __forceinline__ float silu_f(float v) { return v * sigm_f(v); }

// B fragment: 32x16 f16, zero-padded from W[fi][fo] (row-major, k*fo+n).
// lane<16: col=lane, K=0..7,16..23 ; lane>=16: col=lane-16, K=8..15,24..31
__device__ __forceinline__ v16h load_w_frag(const float* __restrict__ W, int fi, int fo) {
  const int lane = threadIdx.x & 31;
  const int n  = lane & 15;
  const int kb = (lane < 16) ? 0 : 8;
  v16h b;
#pragma unroll
  for (int i = 0; i < 16; ++i) {
    const int k = kb + i + ((i < 8) ? 0 : 8);
    const float v = (k < fi && n < fo) ? W[k * fo + n] : 0.0f;
    b[i] = (_Float16)v;
  }
  return b;
}

// A fragment from LDS tile [16 rows][32 K] f16.
// lane<16: row=lane,    K = 0..7,  16..23
// lane>=16: row=lane-16, K = 8..15, 24..31
__device__ __forceinline__ v16h load_a_frag32(const _Float16* __restrict__ s) {
  const int lane = threadIdx.x & 31;
  const int m  = lane & 15;
  const int kb = (lane < 16) ? 0 : 8;
  v16h a;
#pragma unroll
  for (int i = 0; i < 8; ++i) a[i]     = s[m * 32 + kb + i];
#pragma unroll
  for (int i = 0; i < 8; ++i) a[i + 8] = s[m * 32 + 16 + kb + i];
  return a;
}

// A fragment from LDS tile [16 rows][16 K] f16 (K>=16 is zero padding).
__device__ __forceinline__ v16h load_a_frag16(const _Float16* __restrict__ s) {
  const int lane = threadIdx.x & 31;
  const int m  = lane & 15;
  const int kb = (lane < 16) ? 0 : 8;
  v16h a;
#pragma unroll
  for (int i = 0; i < 8; ++i) a[i]     = s[m * 16 + kb + i];
#pragma unroll
  for (int i = 0; i < 8; ++i) a[i + 8] = (_Float16)0.0f;
  return a;
}

__device__ __forceinline__ v8f splat8(float v) {
  v8f c;
#pragma unroll
  for (int r = 0; r < 8; ++r) c[r] = v;
  return c;
}

// D = A*B + C  (bias pre-loaded into C)
__device__ __forceinline__ v8f wmma_f16f32(v16h a, v16h b, v8f c) {
  return __builtin_amdgcn_wmma_f32_16x16x32_f16(false, a, false, b,
                                                (short)0, c, false, false);
}

// ---------------------------------------------------------------------------
// Edge kernel: per 16-edge tile -> feat[16x27] -> m=silu(silu(feat W1)W2),
// attn=sigmoid(m Wa), msg_h=attn*m ; phi path identical shape ; scatter-add.
// ---------------------------------------------------------------------------
__launch_bounds__(256)
__global__ void egnn_edge(
    const float* __restrict__ h, const float* __restrict__ x,
    const int* __restrict__ src, const int* __restrict__ dst,
    const float* __restrict__ ea,
    const float* __restrict__ Wm1, const float* __restrict__ bm1,
    const float* __restrict__ Wm2, const float* __restrict__ bm2,
    const float* __restrict__ Wat, const float* __restrict__ bat,
    const float* __restrict__ Wx1, const float* __restrict__ bx1,
    const float* __restrict__ Wx2, const float* __restrict__ bx2,
    const float* __restrict__ Wx3, const float* __restrict__ bx3,
    float* __restrict__ aggh, float* __restrict__ aggx, int E) {
  __shared__ _Float16 s_feat[8][16][32];
  __shared__ _Float16 s_mid [8][16][16];
  __shared__ float    s_m2  [8][16][16];
  __shared__ float    s_phi [8][16][16];
  __shared__ float    s_geo [8][16][4];

  const int wave = threadIdx.x >> 5;
  const int lane = threadIdx.x & 31;
  const int q    = lane & 15;
  const int e0   = (blockIdx.x * 8 + wave) * 16;
  const int eg   = e0 + q;
  const bool ev  = eg < E;
  const int e    = ev ? eg : (E - 1);       // clamped: loads always in-bounds
  const float sel = ev ? 1.0f : 0.0f;       // mask instead of branches

  // Resident B fragments (weights are tiny; L2-resident scalar loads)
  const v16h Bm1 = load_w_frag(Wm1, 27, 11);
  const v16h Bm2 = load_w_frag(Wm2, 11, 11);
  const v16h Bx1 = load_w_frag(Wx1, 27, 11);
  const v16h Bx2 = load_w_frag(Wx2, 11, 11);

  const int  n  = q;                        // output feature held by this lane
  const int  nc = (n < 11) ? n : 10;        // clamped: unconditional bias loads
  const float zf = (n < 11) ? 1.0f : 0.0f;  // invalid columns zeroed post-WMMA
  const float bias_m1 = bm1[nc], bias_m2 = bm2[nc];
  const float bias_x1 = bx1[nc], bias_x2 = bx2[nc];

  const int si = src[e];
  const int di = dst[e];
  const float dx = x[di * 3 + 0] - x[si * 3 + 0];
  const float dy = x[di * 3 + 1] - x[si * 3 + 1];
  const float dz = x[di * 3 + 2] - x[si * 3 + 2];
  const float d2 = dx * dx + dy * dy + dz * dz;
  if (lane < 16) {
    s_geo[wave][q][0] = sqrtf(d2);
    s_geo[wave][q][1] = dx;
    s_geo[wave][q][2] = dy;
    s_geo[wave][q][3] = dz;
  }
  // feat = [h_i(11), h_j(11), d^2, edge_attr(4), pad] ; lane pair (q,q+16)
  const int kb = (lane < 16) ? 0 : 16;
#pragma unroll
  for (int i = 0; i < 16; ++i) {
    const int k = kb + i;        // lane-dependent through kb (0 or 16)
    float v;
    if (k < 11)       v = h[di * 11 + k];
    else if (k < 22)  v = h[si * 11 + (k - 11)];
    else if (k == 22) v = d2;
    else if (k < 27)  v = ea[e * 4 + (k - 23)];
    else              v = 0.0f;
    s_feat[wave][q][k] = (_Float16)(sel * v);
  }
  __syncthreads();

  const v16h Af   = load_a_frag32(&s_feat[wave][0][0]);
  const int  mrow = (lane < 16) ? 0 : 8;     // edge-row group

  { // m1 = silu(feat @ Wm1 + bm1)
    v8f acc = wmma_f16f32(Af, Bm1, splat8(bias_m1));
#pragma unroll
    for (int r = 0; r < 8; ++r)
      s_mid[wave][mrow + r][n] = (_Float16)(zf * silu_f(acc[r]));
  }
  __syncthreads();
  { // m2 = silu(m1 @ Wm2 + bm2)
    const v16h Am = load_a_frag16(&s_mid[wave][0][0]);
    v8f acc = wmma_f16f32(Am, Bm2, splat8(bias_m2));
#pragma unroll
    for (int r = 0; r < 8; ++r)
      s_m2[wave][mrow + r][n] = zf * silu_f(acc[r]);
  }
  __syncthreads();
  { // phi1 = silu(feat @ Wx1 + bx1)   (reuse s_mid)
    v8f acc = wmma_f16f32(Af, Bx1, splat8(bias_x1));
#pragma unroll
    for (int r = 0; r < 8; ++r)
      s_mid[wave][mrow + r][n] = (_Float16)(zf * silu_f(acc[r]));
  }
  __syncthreads();
  { // phi2 = silu(phi1 @ Wx2 + bx2)
    const v16h Ap = load_a_frag16(&s_mid[wave][0][0]);
    v8f acc = wmma_f16f32(Ap, Bx2, splat8(bias_x2));
#pragma unroll
    for (int r = 0; r < 8; ++r)
      s_phi[wave][mrow + r][n] = zf * silu_f(acc[r]);
  }
  __syncthreads();

  // Scatter: lanes 0-15 -> msg_h (attn * m2), lanes 16-31 -> msg_x.
  if (ev) {
    if (lane < 16) {
      float a = bat[0];
#pragma unroll
      for (int f = 0; f < 11; ++f) a += s_m2[wave][q][f] * Wat[f];
      a = sigm_f(a);
#pragma unroll
      for (int f = 0; f < 11; ++f)
        atomicAdd(&aggh[di * 11 + f], a * s_m2[wave][q][f]);
    } else {
      float p = bx3[0];
#pragma unroll
      for (int f = 0; f < 11; ++f) p += s_phi[wave][q][f] * Wx3[f];
      const float d = s_geo[wave][q][0];
      const float c = p * fast_rcp(d + 1.0f);
      atomicAdd(&aggx[di * 3 + 0], c * s_geo[wave][q][1]);
      atomicAdd(&aggx[di * 3 + 1], c * s_geo[wave][q][2]);
      atomicAdd(&aggx[di * 3 + 2], c * s_geo[wave][q][3]);
    }
  }
}

// ---------------------------------------------------------------------------
// Node kernel: upd = silu([h,agg_h] @ Wu1 + b) @ Wu2 + b ; h += upd ; x += agg_x
// ---------------------------------------------------------------------------
__launch_bounds__(256)
__global__ void egnn_node(
    float* __restrict__ h, float* __restrict__ x,
    const float* __restrict__ aggh, const float* __restrict__ aggx,
    const float* __restrict__ Wu1, const float* __restrict__ bu1,
    const float* __restrict__ Wu2, const float* __restrict__ bu2, int N) {
  __shared__ _Float16 s_feat[8][16][32];
  __shared__ _Float16 s_mid [8][16][16];

  const int wave = threadIdx.x >> 5;
  const int lane = threadIdx.x & 31;
  const int q    = lane & 15;
  const int i0   = (blockIdx.x * 8 + wave) * 16;
  const int ig   = i0 + q;
  const bool iv  = ig < N;
  const int i    = iv ? ig : (N - 1);
  const float sel = iv ? 1.0f : 0.0f;

  const v16h Bu1 = load_w_frag(Wu1, 22, 11);
  const v16h Bu2 = load_w_frag(Wu2, 11, 11);

  const int  n  = q;
  const int  nc = (n < 11) ? n : 10;
  const float zf = (n < 11) ? 1.0f : 0.0f;
  const float bias_u1 = bu1[nc], bias_u2 = bu2[nc];

  const int kb = (lane < 16) ? 0 : 16;
#pragma unroll
  for (int t = 0; t < 16; ++t) {
    const int k = kb + t;
    float v;
    if (k < 11)      v = h[i * 11 + k];
    else if (k < 22) v = aggh[i * 11 + (k - 11)];
    else             v = 0.0f;
    s_feat[wave][q][k] = (_Float16)(sel * v);
  }
  if (lane < 16 && iv) {
#pragma unroll
    for (int c = 0; c < 3; ++c) x[i * 3 + c] += aggx[i * 3 + c];
  }
  __syncthreads();

  const v16h Af   = load_a_frag32(&s_feat[wave][0][0]);
  const int  mrow = (lane < 16) ? 0 : 8;

  {
    v8f acc = wmma_f16f32(Af, Bu1, splat8(bias_u1));
#pragma unroll
    for (int r = 0; r < 8; ++r)
      s_mid[wave][mrow + r][n] = (_Float16)(zf * silu_f(acc[r]));
  }
  __syncthreads();
  {
    const v16h Am = load_a_frag16(&s_mid[wave][0][0]);
    v8f acc = wmma_f16f32(Am, Bu2, splat8(bias_u2));
    if (n < 11) {
#pragma unroll
      for (int r = 0; r < 8; ++r) {
        const int node = i0 + mrow + r;
        if (node < N) h[node * 11 + n] += acc[r];   // residual (bias in C)
      }
    }
  }
}

// ---------------------------------------------------------------------------
// Utility kernels
// ---------------------------------------------------------------------------
__global__ void zero_f32(float* __restrict__ p, int n) {
  const int i = blockIdx.x * blockDim.x + threadIdx.x;
  if (i < n) p[i] = 0.0f;
}
__global__ void copy_f32(float* __restrict__ d, const float* __restrict__ s, int n) {
  const int i = blockIdx.x * blockDim.x + threadIdx.x;
  if (i < n) d[i] = s[i];
}
__global__ void pool_kernel(const float* __restrict__ h, const int* __restrict__ batch,
                            float* __restrict__ gsum, int N) {
  const int i = blockIdx.x * blockDim.x + threadIdx.x;
  if (i >= N) return;
  const int g = batch[i];
#pragma unroll
  for (int f = 0; f < 11; ++f) atomicAdd(&gsum[g * 12 + f], h[i * 11 + f]);
  atomicAdd(&gsum[g * 12 + 11], 1.0f);
}
__global__ void head_kernel(const float* __restrict__ gsum, const float* __restrict__ Wp,
                            const float* __restrict__ bp, float* __restrict__ out, int G) {
  const int g = blockIdx.x * blockDim.x + threadIdx.x;
  if (g >= G) return;
  const float inv = fast_rcp(fmaxf(gsum[g * 12 + 11], 1.0f));
  float acc = bp[0];
#pragma unroll
  for (int f = 0; f < 11; ++f) acc += (gsum[g * 12 + f] * inv) * Wp[f];
  out[g] = acc;
}

// ---------------------------------------------------------------------------
// Host launcher
// ---------------------------------------------------------------------------
extern "C" void kernel_launch(void* const* d_in, const int* in_sizes, int n_in,
                              void* d_out, int out_size, void* d_ws, size_t ws_size,
                              hipStream_t stream) {
  const float* h_in = (const float*)d_in[0];
  const float* pos  = (const float*)d_in[1];
  const int*   eix  = (const int*)d_in[2];
  const float* ea   = (const float*)d_in[3];
  const int*   batch= (const int*)d_in[4];
  const float* Wm1  = (const float*)d_in[5];
  const float* bm1  = (const float*)d_in[6];
  const float* Wm2  = (const float*)d_in[7];
  const float* bm2  = (const float*)d_in[8];
  const float* Wat  = (const float*)d_in[9];
  const float* bat  = (const float*)d_in[10];
  const float* Wu1  = (const float*)d_in[11];
  const float* bu1  = (const float*)d_in[12];
  const float* Wu2  = (const float*)d_in[13];
  const float* bu2  = (const float*)d_in[14];
  const float* Wx1  = (const float*)d_in[15];
  const float* bx1  = (const float*)d_in[16];
  const float* Wx2  = (const float*)d_in[17];
  const float* bx2  = (const float*)d_in[18];
  const float* Wx3  = (const float*)d_in[19];
  const float* bx3  = (const float*)d_in[20];
  const float* Wp   = (const float*)d_in[21];
  const float* bp   = (const float*)d_in[22];
  (void)n_in; (void)ws_size;

  const int N = in_sizes[0] / 11;
  const int E = in_sizes[2] / 2;
  const int G = out_size;
  const int* src = eix;
  const int* dst = eix + E;

  float* ws   = (float*)d_ws;
  float* h    = ws;                               // N*11
  float* x    = h    + (size_t)N * 11;            // N*3
  float* aggh = x    + (size_t)N * 3;             // N*11
  float* aggx = aggh + (size_t)N * 11;            // N*3
  float* gsum = aggx + (size_t)N * 3;             // G*12

  const int T = 256;
  copy_f32<<<(N * 11 + T - 1) / T, T, 0, stream>>>(h, h_in, N * 11);
  copy_f32<<<(N * 3  + T - 1) / T, T, 0, stream>>>(x, pos, N * 3);

  const int eblocks = ((E + 15) / 16 + 7) / 8;
  const int nblocks = ((N + 15) / 16 + 7) / 8;
  const int S27 = 27 * 11, S11 = 11 * 11, S22 = 22 * 11;

  for (int l = 0; l < 4; ++l) {
    zero_f32<<<(N * 11 + T - 1) / T, T, 0, stream>>>(aggh, N * 11);
    zero_f32<<<(N * 3  + T - 1) / T, T, 0, stream>>>(aggx, N * 3);
    egnn_edge<<<eblocks, T, 0, stream>>>(
        h, x, src, dst, ea,
        Wm1 + l * S27, bm1 + l * 11, Wm2 + l * S11, bm2 + l * 11,
        Wat + l * 11,  bat + l,
        Wx1 + l * S27, bx1 + l * 11, Wx2 + l * S11, bx2 + l * 11,
        Wx3 + l * 11,  bx3 + l,
        aggh, aggx, E);
    egnn_node<<<nblocks, T, 0, stream>>>(
        h, x, aggh, aggx,
        Wu1 + l * S22, bu1 + l * 11, Wu2 + l * S11, bu2 + l * 11, N);
  }

  zero_f32<<<(G * 12 + T - 1) / T, T, 0, stream>>>(gsum, G * 12);
  pool_kernel<<<(N + T - 1) / T, T, 0, stream>>>(h, batch, gsum, N);
  head_kernel<<<(G + T - 1) / T, T, 0, stream>>>(gsum, Wp, bp, (float*)d_out, G);
}
